// WUM_34823594836499
// MI455X (gfx1250) — matrix operand order repaired and checked by reading, not compiled, not run on hardware
//
#include <hip/hip_runtime.h>
#include <cmath>

typedef __attribute__((ext_vector_type(16))) _Float16 v16h;
typedef __attribute__((ext_vector_type(8)))  float    v8f;

static constexpr int BATCH = 2;
static constexpr int CH    = 64;
static constexpr int IMH   = 256;
static constexpr int IMW   = 256;
static constexpr int HWPX  = IMH * IMW;          // 65536
static constexpr int QKVC  = 192;                // 3*CH
static constexpr int HIDC  = 170;                // int(64*2.66)
static constexpr int NWIN  = (IMH / 8) * (IMW / 8); // 1024
static constexpr size_t NCHW = (size_t)BATCH * CH * HWPX;

// ---- WMMA lane->element maps (CDNA5 16-bit 16x16x32 layouts) -------------
__device__ __forceinline__ int kmapA(int e, int lane) {
  return ((e >> 3) << 4) + ((lane >> 4) << 3) + (e & 7);
}
__device__ __forceinline__ int kmapB(int e, int lane) {
  return ((e >> 3) << 4) + ((lane >> 4) << 3) + (e & 7);
}
__device__ __forceinline__ int iclamp(int v, int lo, int hi) {
  return v < lo ? lo : (v > hi ? hi : v);
}
// pack two contiguous float4 into 8 f16 lanes of a v16h fragment
__device__ __forceinline__ void cvt8(const float4 a, const float4 b, v16h& v, int base) {
  v[base + 0] = (_Float16)a.x; v[base + 1] = (_Float16)a.y;
  v[base + 2] = (_Float16)a.z; v[base + 3] = (_Float16)a.w;
  v[base + 4] = (_Float16)b.x; v[base + 5] = (_Float16)b.y;
  v[base + 6] = (_Float16)b.z; v[base + 7] = (_Float16)b.w;
}

#define WMMA16(a, b, c) \
  __builtin_amdgcn_wmma_f32_16x16x32_f16(false, (a), false, (b), (short)0, (c), false, false)

// ---- conv1x1 as GEMM with LDS-staged activation tile ---------------------
// One block = GW o-tiles x 1 p-tile. Block stages [KPAD x 16] activations
// into LDS once (zero padded), each wave streams its weight rows and WMMAs.
template <int ICT, int OCT>
__global__ __launch_bounds__(256)
void conv1x1_wmma_kernel(const float* __restrict__ in, const float* __restrict__ w,
                         const float* __restrict__ res, float* __restrict__ out) {
  constexpr int OCTILES = (OCT + 15) / 16;
  constexpr int GW = OCTILES < 8 ? OCTILES : 8;       // waves per block
  constexpr int OGROUPS = (OCTILES + GW - 1) / GW;
  constexpr int PT = HWPX / 16;
  constexpr int KPAD = (ICT + 31) & ~31;
  __shared__ float lb[KPAD * 17];                      // [k][pix], stride 17
  int t = threadIdx.x;
  int blk = blockIdx.x;
  int pt = blk % PT;
  int og = (blk / PT) % OGROUPS;
  int b  = blk / (PT * OGROUPS);
  int p0 = pt * 16;

  const float* bsrc = in + (size_t)b * ICT * HWPX + p0;
  for (int idx = t; idx < KPAD * 16; idx += GW * 32) {
    int k = idx >> 4, pix = idx & 15;
    float v = bsrc[(size_t)iclamp(k, 0, ICT - 1) * HWPX + pix];
    lb[k * 17 + pix] = (k < ICT) ? v : 0.f;
  }
  __syncthreads();

  int wave = t >> 5, lane = t & 31;
  int ot = og * GW + wave;
  if (ot >= OCTILES) return;
  int o0 = ot * 16;
  int m = lane & 15;
  int khalf = (lane >> 4) << 3;
  int kqtr  = (lane >> 4) << 4;
  int orow = o0 + m;
  bool ovalid = orow < OCT;
  const float* wrow = w + (size_t)(ovalid ? orow : OCT - 1) * ICT;

  v8f acc = {};
  constexpr int KFULL = ICT & ~31;
#pragma unroll
  for (int k0 = 0; k0 < KFULL; k0 += 32) {
    v16h av, bv;
    if constexpr ((ICT & 3) == 0) {                    // aligned rows: b128 loads
      float4 a0 = *(const float4*)(wrow + k0 + khalf);
      float4 a1 = *(const float4*)(wrow + k0 + khalf + 4);
      float4 a2 = *(const float4*)(wrow + k0 + 16 + khalf);
      float4 a3 = *(const float4*)(wrow + k0 + 16 + khalf + 4);
      cvt8(a0, a1, av, 0); cvt8(a2, a3, av, 8);
    } else {
#pragma unroll
      for (int e = 0; e < 16; ++e)
        av[e] = (_Float16)wrow[k0 + ((e >> 3) << 4) + khalf + (e & 7)];
    }
    if (!ovalid) {
#pragma unroll
      for (int e = 0; e < 16; ++e) av[e] = (_Float16)0.f;
    }
#pragma unroll
    for (int e = 0; e < 16; ++e)
      bv[e] = (_Float16)lb[(k0 + ((e >> 3) << 4) + khalf + (e & 7)) * 17 + m];
    acc = WMMA16(av, bv, acc);
  }
  if constexpr ((ICT & 31) != 0) {                     // ragged K tail (IC=170)
    constexpr int k0 = KFULL;
    v16h av, bv;
#pragma unroll
    for (int e = 0; e < 16; ++e) {
      int ka = k0 + ((e >> 3) << 4) + khalf + (e & 7);
      float aval = wrow[iclamp(ka, 0, ICT - 1)];
      av[e] = (_Float16)((ovalid && ka < ICT) ? aval : 0.f);
      bv[e] = (_Float16)lb[(k0 + ((e >> 3) << 4) + khalf + (e & 7)) * 17 + m]; // LDS already zero padded
    }
    acc = WMMA16(av, bv, acc);
  }

  int n = lane & 15, half = lane >> 4;
  int ob = o0 + 8 * half;
  size_t boff = ((size_t)b * OCT + ob) * HWPX + p0 + n;
  float* dst = out + boff;
  const float* rsp = res ? res + boff : nullptr;
#pragma unroll
  for (int r = 0; r < 8; ++r) {
    if (ob + r < OCT) {
      float v = acc[r];
      if (rsp) v += rsp[(size_t)r * HWPX];
      dst[(size_t)r * HWPX] = v;                       // immediate-offset stores
    }
  }
}

// ---- LayerNorm over channel dim (biased var, eps=1e-5), single pass ------
__global__ void ln_kernel(const float* __restrict__ in, const float* __restrict__ g,
                          const float* __restrict__ beta, float* __restrict__ out) {
  int i = blockIdx.x * blockDim.x + threadIdx.x;
  if (i >= BATCH * HWPX) return;
  int b = i >> 16, p = i & (HWPX - 1);
  const float* xp = in + (size_t)b * CH * HWPX + p;
  float v[CH];
  float s = 0.f, s2 = 0.f;
#pragma unroll
  for (int c = 0; c < CH; ++c) {
    v[c] = xp[(size_t)c * HWPX];
    s += v[c];
    s2 += v[c] * v[c];
  }
  float mu = s * (1.f / CH);
  float var = s2 * (1.f / CH) - mu * mu;
  float inv = rsqrtf(fmaxf(var, 0.f) + 1e-5f);
  float* op = out + (size_t)b * CH * HWPX + p;
#pragma unroll
  for (int c = 0; c < CH; ++c)
    op[(size_t)c * HWPX] = (v[c] - mu) * inv * g[c] + beta[c];
}

// ---- cyclic roll by +sh on both spatial axes (reads shifted) -------------
__global__ void roll_kernel(const float* __restrict__ in, float* __restrict__ out,
                            int nch, int sh) {
  size_t i = (size_t)blockIdx.x * blockDim.x + threadIdx.x;
  size_t tot = (size_t)BATCH * nch * HWPX;
  if (i >= tot) return;
  int p = (int)(i % HWPX);
  size_t bc = i / HWPX;
  int y = p / IMW, x = p % IMW;
  int ys = (y + sh) & (IMH - 1), xs = (x + sh) & (IMW - 1);
  out[i] = in[bc * HWPX + ys * IMW + xs];
}

// ---- depthwise 3x3, SAME zero pad (branchless) ---------------------------
__global__ void dwconv3_kernel(const float* __restrict__ in, const float* __restrict__ w,
                               float* __restrict__ out, int nch) {
  size_t i = (size_t)blockIdx.x * blockDim.x + threadIdx.x;
  size_t tot = (size_t)BATCH * nch * HWPX;
  if (i >= tot) return;
  int p = (int)(i % HWPX);
  size_t bc = i / HWPX;
  int c = (int)(bc % nch);
  int y = p / IMW, x = p % IMW;
  const float* wp = w + c * 9;
  const float* ip = in + bc * HWPX;
  float s = 0.f;
#pragma unroll
  for (int dy = -1; dy <= 1; ++dy) {
    int yy = y + dy;
    bool yv = (unsigned)yy < (unsigned)IMH;
    int yyc = iclamp(yy, 0, IMH - 1);
#pragma unroll
    for (int dx = -1; dx <= 1; ++dx) {
      int xx = x + dx;
      bool xv = (unsigned)xx < (unsigned)IMW;
      int xxc = iclamp(xx, 0, IMW - 1);
      float val = ip[yyc * IMW + xxc];
      s += (yv && xv) ? wp[(dy + 1) * 3 + (dx + 1)] * val : 0.f;
    }
  }
  out[i] = s;
}

__global__ void add_kernel(const float* __restrict__ a, const float* __restrict__ b,
                           float* __restrict__ o) {
  size_t i = (size_t)blockIdx.x * blockDim.x + threadIdx.x;
  if (i >= NCHW) return;
  o[i] = a[i] + b[i];
}

// ---- FFN gate: g = gelu(h[0:HID]) * h[HID:2HID] (exact gelu) -------------
__global__ void gate_kernel(const float* __restrict__ h, float* __restrict__ g) {
  size_t i = (size_t)blockIdx.x * blockDim.x + threadIdx.x;
  size_t tot = (size_t)BATCH * HIDC * HWPX;
  if (i >= tot) return;
  int p = (int)(i % HWPX);
  size_t bc = i / HWPX;
  int c = (int)(bc % HIDC);
  int b = (int)(bc / HIDC);
  float x1 = h[((size_t)(b * 2 * HIDC + c)) * HWPX + p];
  float x2 = h[((size_t)(b * 2 * HIDC + HIDC + c)) * HWPX + p];
  float ge = 0.5f * x1 * (1.f + erff(x1 * 0.70710678118654752f));
  g[i] = ge * x2;
}

// ---- RSA: attn[c,d] = relu(temp * sum_p qn[p,c]*kn[p,d]) per 8x8 window --
__global__ __launch_bounds__(256)
void rsa_attn_kernel(const float* __restrict__ qkv, const float* __restrict__ tp,
                     float* __restrict__ attn) {
  __shared__ float qs[64][65];
  __shared__ float ks[64][65];
  int wid = blockIdx.x;
  int b = wid >> 10, win = wid & 1023;
  int wy = win >> 5, wx = win & 31;
  int t = threadIdx.x;
  for (int idx = t; idx < 4096; idx += 256) {
    int p = idx >> 6, c = idx & 63;
    int pix = (wy * 8 + (p >> 3)) * IMW + wx * 8 + (p & 7);
    qs[p][c] = qkv[((size_t)(b * QKVC + c)) * HWPX + pix];
    ks[p][c] = qkv[((size_t)(b * QKVC + 64 + c)) * HWPX + pix];
  }
  __syncthreads();
  // L2-normalize each pixel row over channels, in place (removes the
  // per-fragment multiply so the WMMA loop is pure ds_load + cvt)
  if (t < 64) {
    float s = 0.f;
#pragma unroll
    for (int c = 0; c < 64; ++c) s += qs[t][c] * qs[t][c];
    float inv = 1.f / fmaxf(sqrtf(s), 1e-12f);
#pragma unroll
    for (int c = 0; c < 64; ++c) qs[t][c] *= inv;
  } else if (t < 128) {
    int p = t - 64;
    float s = 0.f;
#pragma unroll
    for (int c = 0; c < 64; ++c) s += ks[p][c] * ks[p][c];
    float inv = 1.f / fmaxf(sqrtf(s), 1e-12f);
#pragma unroll
    for (int c = 0; c < 64; ++c) ks[p][c] *= inv;
  }
  __syncthreads();
  int wave = t >> 5, lane = t & 31;
  int mn = lane & 15, half = lane >> 4;
  float tv = tp[0];
  for (int pair = wave; pair < 16; pair += 8) {
    int ct = pair >> 2, dt = pair & 3;
    v8f acc = {};
#pragma unroll
    for (int k0 = 0; k0 < 64; k0 += 32) {
      v16h av, bv;
#pragma unroll
      for (int e = 0; e < 16; ++e) av[e] = (_Float16)qs[k0 + kmapA(e, lane)][ct * 16 + mn];
#pragma unroll
      for (int e = 0; e < 16; ++e) bv[e] = (_Float16)ks[k0 + kmapB(e, lane)][dt * 16 + mn];
      acc = WMMA16(av, bv, acc);
    }
    float* dst = attn + (((size_t)(b * NWIN + win)) * 64 + ct * 16 + 8 * half) * 64 + dt * 16 + mn;
#pragma unroll
    for (int r = 0; r < 8; ++r) {
      float v = acc[r] * tv;
      dst[(size_t)r * 64] = v > 0.f ? v : 0.f;
    }
  }
}

// ---- RSA apply: out[p,d] = sum_c v[p,c]*attn[c,d]; coalesced unroll ------
__global__ __launch_bounds__(256)
void rsa_apply_kernel(const float* __restrict__ src, int srcC, int srcOff, int fold,
                      const float* __restrict__ attn, float* __restrict__ y) {
  __shared__ float vs[64][65];
  __shared__ float as[64][65];
  int wid = blockIdx.x;
  int b = wid >> 10, win = wid & 1023;
  int wy = win >> 5, wx = win & 31;
  int t = threadIdx.x;
  for (int idx = t; idx < 4096; idx += 256) {
    int p = idx >> 6, c = idx & 63;
    int ys = wy * 8 + (p >> 3), xs = wx * 8 + (p & 7);
    int yy = fold ? ((ys + 4) & (IMH - 1)) : ys;
    int xx = fold ? ((xs + 4) & (IMW - 1)) : xs;
    vs[p][c] = src[((size_t)(b * srcC + srcOff + c)) * HWPX + yy * IMW + xx];
    as[p][c] = attn[(((size_t)(b * NWIN + win)) * 64 + p) * 64 + c];
  }
  __syncthreads();
  int wave = t >> 5, lane = t & 31;
  int mn = lane & 15, half = lane >> 4;
  v8f accs[2];
#pragma unroll
  for (int pi = 0; pi < 2; ++pi) {
    int pair = wave + pi * 8;
    int ptile = pair >> 2, dt = pair & 3;
    v8f acc = {};
#pragma unroll
    for (int k0 = 0; k0 < 64; k0 += 32) {
      v16h av, bv;
#pragma unroll
      for (int e = 0; e < 16; ++e) av[e] = (_Float16)vs[ptile * 16 + mn][k0 + kmapA(e, lane)];
#pragma unroll
      for (int e = 0; e < 16; ++e) bv[e] = (_Float16)as[k0 + kmapB(e, lane)][dt * 16 + mn];
      acc = WMMA16(av, bv, acc);
    }
    accs[pi] = acc;
  }
  __syncthreads();                 // everyone done reading as[]
#pragma unroll
  for (int pi = 0; pi < 2; ++pi) {
    int pair = wave + pi * 8;
    int ptile = pair >> 2, dt = pair & 3;
#pragma unroll
    for (int r = 0; r < 8; ++r)
      as[ptile * 16 + 8 * half + r][dt * 16 + mn] = accs[pi][r];
  }
  __syncthreads();
  // coalesced store with roll-back fold: consecutive threads -> consecutive x
  for (int idx = t; idx < 4096; idx += 256) {
    int d = idx >> 6, p = idx & 63;
    int ys = wy * 8 + (p >> 3), xs = wx * 8 + (p & 7);
    int yo = (ys + 4) & (IMH - 1), xo = (xs + 4) & (IMW - 1);
    y[((size_t)(b * CH + d)) * HWPX + yo * IMW + xo] = as[p][d];
  }
}

// ---- GSA: L2 norms over n per (b, q/k channel) ---------------------------
__global__ void gsa_norms_kernel(const float* __restrict__ qkv, float* __restrict__ norms) {
  __shared__ float red[256];
  int row = blockIdx.x;                 // [0, BATCH*128)
  int b = row >> 7, ch = row & 127;
  const float* p = qkv + ((size_t)(b * QKVC + ch)) * HWPX;
  float s = 0.f;
  for (int i = threadIdx.x; i < HWPX; i += 256) { float v = p[i]; s += v * v; }
  red[threadIdx.x] = s;
  __syncthreads();
  for (int st = 128; st > 0; st >>= 1) {
    if ((int)threadIdx.x < st) red[threadIdx.x] += red[threadIdx.x + st];
    __syncthreads();
  }
  if (threadIdx.x == 0) norms[row] = sqrtf(red[0]);
}

// ---- GSA raw scores: per-(b,h) 16x16, K=65536 in 64 chunks, b128 loads ---
__global__ __launch_bounds__(256)
void gsa_raw_kernel(const float* __restrict__ qkv, float* __restrict__ part) {
  int gw = (int)((blockIdx.x * blockDim.x + threadIdx.x) >> 5);
  int lane = threadIdx.x & 31;
  int chunk = gw & 63;
  int bh = gw >> 6;
  if (bh >= BATCH * 4) return;
  int b = bh >> 2, h = bh & 3;
  int m = lane & 15;
  int khalf = (lane >> 4) << 3;
  int kqtr  = (lane >> 4) << 4;
  const float* qrow = qkv + ((size_t)(b * QKVC + h * 16 + m)) * HWPX + chunk * 1024;
  const float* krow = qkv + ((size_t)(b * QKVC + 64 + h * 16 + m)) * HWPX + chunk * 1024;
  v8f acc = {};
  for (int kk = 0; kk < 1024; kk += 32) {
    const float* qp = qrow + kk + khalf;
    const float* kp = krow + kk + khalf;
    float4 a0 = *(const float4*)(qp);
    float4 a1 = *(const float4*)(qp + 4);
    float4 a2 = *(const float4*)(qp + 16);
    float4 a3 = *(const float4*)(qp + 20);
    float4 b0 = *(const float4*)(kp);
    float4 b1 = *(const float4*)(kp + 4);
    float4 b2 = *(const float4*)(kp + 16);
    float4 b3 = *(const float4*)(kp + 20);
    v16h av, bv;
    cvt8(a0, a1, av, 0); cvt8(a2, a3, av, 8);
    cvt8(b0, b1, bv, 0); cvt8(b2, b3, bv, 8);
    acc = WMMA16(av, bv, acc);
  }
  int n = lane & 15, half = lane >> 4;
  float* dst = part + ((size_t)(bh * 64 + chunk)) * 256 + (8 * half) * 16 + n;
#pragma unroll
  for (int r = 0; r < 8; ++r) dst[r * 16] = acc[r];    // deterministic partials
}

__global__ void gsa_fin_kernel(const float* __restrict__ part, const float* __restrict__ norms,
                               const float* __restrict__ tp, float* __restrict__ attn) {
  int i = blockIdx.x * blockDim.x + threadIdx.x;
  if (i >= BATCH * 4 * 256) return;
  int bh = i >> 8, cd = i & 255;
  int b = bh >> 2, h = bh & 3;
  int c = cd >> 4, d = cd & 15;
  float s = 0.f;
  for (int ch = 0; ch < 64; ++ch) s += part[((size_t)(bh * 64 + ch)) * 256 + cd];
  float nq = fmaxf(norms[b * 128 + h * 16 + c], 1e-12f);
  float nk = fmaxf(norms[b * 128 + 64 + h * 16 + d], 1e-12f);
  float v = tp[0] * s / (nq * nk);
  attn[i] = v > 0.f ? v : 0.f;
}

// ---- GSA apply + head-merge (channel = c*HEADS + h) ----------------------
__global__ void gsa_apply_kernel(const float* __restrict__ src, int srcC, int srcOff,
                                 const float* __restrict__ attn, float* __restrict__ y) {
  __shared__ float at[1024];
  int b = blockIdx.y;
  int n = blockIdx.x * blockDim.x + threadIdx.x;
  for (int i = threadIdx.x; i < 1024; i += blockDim.x) at[i] = attn[b * 1024 + i];
  __syncthreads();
  if (n >= HWPX) return;
  for (int h = 0; h < 4; ++h) {
    float vv[16];
#pragma unroll
    for (int d = 0; d < 16; ++d)
      vv[d] = src[((size_t)(b * srcC + srcOff + h * 16 + d)) * HWPX + n];
#pragma unroll
    for (int c = 0; c < 16; ++c) {
      float s = 0.f;
#pragma unroll
      for (int d = 0; d < 16; ++d) s += at[(h * 16 + c) * 16 + d] * vv[d];
      y[((size_t)(b * CH + c * 4 + h)) * HWPX + n] = s;
    }
  }
}

// ==========================================================================
extern "C" void kernel_launch(void* const* d_in, const int* in_sizes, int n_in,
                              void* d_out, int out_size, void* d_ws, size_t ws_size,
                              hipStream_t stream) {
  (void)in_sizes; (void)n_in; (void)out_size; (void)ws_size;
  const float* x_in = (const float*)d_in[0];
  const float* S[11];
  const float* Cc[11];
  for (int j = 0; j < 11; ++j) { S[j] = (const float*)d_in[1 + j]; Cc[j] = (const float*)d_in[12 + j]; }
  const size_t LS[11] = {64, 64, 64, 64, (size_t)QKVC * CH, (size_t)QKVC * 9,
                         (size_t)CH * CH, 1, (size_t)2 * HIDC * CH, (size_t)2 * HIDC * 9,
                         (size_t)CH * HIDC};

  float* base = (float*)d_ws;
  size_t off = 0;
  auto take = [&](size_t nelem) { float* p = base + off; off += (nelem + 63) & ~(size_t)63; return p; };
  float* X    = (float*)d_out;                    // running activation lives in d_out
  float* XN   = take(NCHW);
  float* XP   = take(NCHW);                       // x_
  float* Y    = take(NCHW);                       // scratch / attn output / xsum
  float* ATTN = take((size_t)BATCH * NWIN * 64 * 64);
  float* BIG1 = take((size_t)BATCH * 2 * HIDC * HWPX);
  float* BIG2 = take((size_t)BATCH * 2 * HIDC * HWPX);
  float* NORM = take(256);
  float* GPRT = take((size_t)BATCH * 4 * 64 * 256);
  float* GATT = take((size_t)BATCH * 4 * 256);

  auto conv = [&](const float* in, const float* w, const float* res, float* out, int IC, int OC) {
    int octiles = (OC + 15) >> 4;
    int gwv = octiles < 8 ? octiles : 8;
    int ogroups = (octiles + gwv - 1) / gwv;
    int blocks = BATCH * ogroups * (HWPX / 16);
    int threads = gwv * 32;
    if (IC == 64 && OC == QKVC)
      conv1x1_wmma_kernel<64, QKVC><<<blocks, threads, 0, stream>>>(in, w, res, out);
    else if (IC == 64 && OC == 2 * HIDC)
      conv1x1_wmma_kernel<64, 2 * HIDC><<<blocks, threads, 0, stream>>>(in, w, res, out);
    else if (IC == HIDC && OC == 64)
      conv1x1_wmma_kernel<HIDC, 64><<<blocks, threads, 0, stream>>>(in, w, res, out);
    else
      conv1x1_wmma_kernel<64, 64><<<blocks, threads, 0, stream>>>(in, w, res, out);
  };
  auto dwconv = [&](const float* in, const float* w, float* out, int nch) {
    dwconv3_kernel<<<(int)(((size_t)BATCH * nch * HWPX) / 256), 256, 0, stream>>>(in, w, out, nch);
  };
  auto ln = [&](const float* in, const float* g, const float* bta, float* out) {
    ln_kernel<<<(BATCH * HWPX) / 256, 256, 0, stream>>>(in, g, bta, out);
  };
  auto add = [&](const float* a, const float* b, float* o) {
    add_kernel<<<(int)(NCHW / 256), 256, 0, stream>>>(a, b, o);
  };

  hipMemcpyAsync(X, x_in, NCHW * sizeof(float), hipMemcpyDeviceToDevice, stream);

  // ---------------- spatial (RSA) branch ----------------
  for (int i = 0; i < 2; ++i) {
    ln(X, S[0] + LS[0] * i, S[1] + LS[1] * i, XN);
    if (i == 0) {
      roll_kernel<<<(int)(NCHW / 256), 256, 0, stream>>>(XN, Y, CH, 4);      // roll -4,-4
      conv(Y, S[4] + LS[4] * i, nullptr, BIG1, CH, QKVC);
      dwconv(BIG1, S[5] + LS[5] * i, BIG2, QKVC);
      rsa_attn_kernel<<<BATCH * NWIN, 256, 0, stream>>>(BIG2, S[7] + i, ATTN);
      rsa_apply_kernel<<<BATCH * NWIN, 256, 0, stream>>>(BIG2, QKVC, 128, 0, ATTN, Y);
    } else {
      rsa_apply_kernel<<<BATCH * NWIN, 256, 0, stream>>>(XN, CH, 0, 1, ATTN, Y);
    }
    conv(Y, S[6] + LS[6] * i, nullptr, XP, CH, CH);                           // x_ = proj
    add(XP, X, Y);                                                            // xsum
    ln(Y, S[2] + LS[2] * i, S[3] + LS[3] * i, XN);
    conv(XN, S[8] + LS[8] * i, nullptr, BIG1, CH, 2 * HIDC);
    dwconv(BIG1, S[9] + LS[9] * i, BIG2, 2 * HIDC);
    gate_kernel<<<(int)(((size_t)BATCH * HIDC * HWPX) / 256), 256, 0, stream>>>(BIG2, BIG1);
    conv(BIG1, S[10] + LS[10] * i, XP, X, HIDC, CH);                          // x = ffd + x_
  }

  // ---------------- channel (GSA) branch ----------------
  for (int i = 0; i < 2; ++i) {
    ln(X, Cc[0] + LS[0] * i, Cc[1] + LS[1] * i, XN);
    if (i == 0) {
      conv(XN, Cc[4] + LS[4] * i, nullptr, BIG1, CH, QKVC);
      dwconv(BIG1, Cc[5] + LS[5] * i, BIG2, QKVC);
      gsa_norms_kernel<<<BATCH * 128, 256, 0, stream>>>(BIG2, NORM);
      gsa_raw_kernel<<<(BATCH * 4 * 64 * 32) / 256, 256, 0, stream>>>(BIG2, GPRT);
      gsa_fin_kernel<<<(BATCH * 4 * 256) / 256, 256, 0, stream>>>(GPRT, NORM, Cc[7] + i, GATT);
      gsa_apply_kernel<<<dim3(HWPX / 256, BATCH), 256, 0, stream>>>(BIG2, QKVC, 128, GATT, Y);
      conv(Y, Cc[6] + LS[6] * i, nullptr, XP, CH, CH);                        // x_ = proj
      add(XP, X, Y);                                                          // xsum = x_ + x
    } else {
      gsa_apply_kernel<<<dim3(HWPX / 256, BATCH), 256, 0, stream>>>(XN, CH, 0, GATT, Y);
      conv(Y, Cc[6] + LS[6] * i, nullptr, X, CH, CH);                         // x overwritten (faithful)
      add(XP, X, Y);                                                          // xsum = stale x_ + new x
    }
    ln(Y, Cc[2] + LS[2] * i, Cc[3] + LS[3] * i, XN);
    conv(XN, Cc[8] + LS[8] * i, nullptr, BIG1, CH, 2 * HIDC);
    dwconv(BIG1, Cc[9] + LS[9] * i, BIG2, 2 * HIDC);
    gate_kernel<<<(int)(((size_t)BATCH * HIDC * HWPX) / 256), 256, 0, stream>>>(BIG2, BIG1);
    conv(BIG1, Cc[10] + LS[10] * i, XP, X, HIDC, CH);                         // final i==1 writes d_out
  }
}
